// LightRecurrentUnitBlock_31198642438609
// MI455X (gfx1250) — matrix operand order repaired and checked by reading, not compiled, not run on hardware
//
#include <hip/hip_runtime.h>
#include <hip/hip_bf16.h>

typedef __bf16 bf16_t;
typedef __attribute__((ext_vector_type(16))) __bf16 v16bf;
typedef __attribute__((ext_vector_type(8)))  float  v8f;
typedef __attribute__((ext_vector_type(4)))  unsigned int u4v;
typedef __attribute__((ext_vector_type(4)))  float  f4v;
typedef __attribute__((ext_vector_type(4)))  unsigned int u32x4;
typedef __attribute__((ext_vector_type(8)))  int i32x8;
typedef __attribute__((ext_vector_type(4)))  int i32x4;

#define DD   512
#define BB   32
#define TT   2048
#define RTOT (BB * TT)   // 65536 rows

__device__ __forceinline__ bf16_t f2bf(float f) {
    unsigned u = __float_as_uint(f);
    unsigned r = (u + 0x7FFFu + ((u >> 16) & 1u)) >> 16;   // RNE
    unsigned short s = (unsigned short)r;
    return __builtin_bit_cast(bf16_t, s);
}

__device__ __forceinline__ v8f wmma_bf16(v16bf a, v16bf b, v8f c) {
    // D(f32 16x16) = A(bf16 16x32) * B(bf16 32x16) + C
    return __builtin_amdgcn_wmma_f32_16x16x32_bf16(
        /*neg_a=*/false, a, /*neg_b=*/false, b,
        /*c_mod=*/(short)0, c, /*reuse_a=*/false, /*reuse_b=*/false);
}

union ABFrag { u4v u[2]; v16bf v; };

// ---------------------------------------------------------------------------
// TDM: DMA a [128 rows x 64 K] bf16 tile (row stride 512 elems) from global
// into LDS (row-major, contiguous 16 KB). 2D descriptor, groups 2/3 zero.
//   Group0: count=1 | lds_addr | global_addr[56:0] | type=2
//   Group1: data_size=2B | tensor_dim0=512 | tensor_dim1=65536 |
//           tile_dim0=64 | tile_dim1=128 | tensor_dim0_stride=512
// ---------------------------------------------------------------------------
__device__ __forceinline__ void tdm_load_tileA(unsigned lds_off, const void* gptr) {
    unsigned long long ga = (unsigned long long)(uintptr_t)gptr;
    u32x4 g0 = { 1u,                                   // count=1, user mode
                 lds_off,                              // lds_addr
                 (unsigned)(ga & 0xFFFFFFFFu),         // global_addr[31:0]
                 (unsigned)((ga >> 32) & 0x01FFFFFFu) | (2u << 30) }; // [56:32] | type=2
    i32x8 g1 = { (int)0x00010000,                      // data_size=1 (2 bytes)
                 (int)(512u << 16),                    // tensor_dim0 = 512 (lo16)
                 0,                                    // dim0 hi16=0, dim1 lo16=0
                 (int)((64u << 16) | 1u),              // dim1 hi16=1 (65536), tile_dim0=64
                 128,                                  // tile_dim1=128, tile_dim2=0
                 512,                                  // tensor_dim0_stride = 512
                 0, 0 };                               // dim1_stride = 0 (unused, 2D)
    i32x4 gz = (i32x4)0;
#if defined(__clang_major__) && (__clang_major__ >= 23)
    __builtin_amdgcn_tensor_load_to_lds(g0, g1, gz, gz, (i32x8)0, 0);
#else
    __builtin_amdgcn_tensor_load_to_lds(g0, g1, gz, gz, 0);
#endif
}

// ---------------------------------------------------------------------------
// Pack W[f32, 512x512 row-major, (k=in_dim, n=out_dim)] into bf16 B-fragment
// order: idx = ((ntile*16 + kc)*32 + lane)*16 + e
//   lane = n%16 + 16*((k%16)>=8) ; e = k%8 + 8*((k%32)>=16) ; kc = k/32 ; ntile = n/16
// ---------------------------------------------------------------------------
__global__ __launch_bounds__(256) void pack_w_kernel(const float* __restrict__ W,
                                                     bf16_t* __restrict__ Wp) {
    int idx = blockIdx.x * 256 + threadIdx.x;      // 0 .. 262143
    int k = idx >> 9, n = idx & 511;
    int ntile = n >> 4, kc = k >> 5, k32 = k & 31;
    int lane = (n & 15) + (((k32 >> 3) & 1) << 4);
    int e = (k32 & 7) + ((k32 >> 4) << 3);
    Wp[(((ntile * 16 + kc) * 32 + lane) << 4) + e] = f2bf(W[idx]);
}

// ---------------------------------------------------------------------------
// RMSNorm: one wave32 per row. Emits bf16 xn (gamma applied) + inv-norm scale.
// ---------------------------------------------------------------------------
__global__ __launch_bounds__(256) void rmsnorm_kernel(const float* __restrict__ x,
                                                      const float* __restrict__ gamma,
                                                      bf16_t* __restrict__ xnb,
                                                      float* __restrict__ rstd) {
    int wave = threadIdx.x >> 5, lane = threadIdx.x & 31;
    size_t row = (size_t)blockIdx.x * 8 + wave;
    const float* xr = x + row * DD;
    float vals[16];
    float ss = 0.f;
#pragma unroll
    for (int i = 0; i < 16; ++i) { float v = xr[lane + 32 * i]; vals[i] = v; ss += v * v; }
#pragma unroll
    for (int off = 16; off >= 1; off >>= 1) ss += __shfl_xor(ss, off, 32);
    float sc = 22.627416997969522f / fmaxf(sqrtf(ss), 1e-12f);   // sqrt(512)/||x||
#pragma unroll
    for (int i = 0; i < 16; ++i) {
        int c = lane + 32 * i;
        xnb[row * DD + c] = f2bf(vals[i] * sc * (gamma[c] + 1.f));
    }
    if (lane == 0) rstd[row] = sc;
}

// ---------------------------------------------------------------------------
// GEMM: out[f32] = act( A[bf16 R x 512] @ Wp[packed bf16] (+bias) (+addend) )
// Block = 256 thr (8 waves), tile 128M x 64N. A tiles DMA'd by the TDM into
// double-buffered row-major LDS (prefetch of chunk k+1 overlaps WMMA on k).
// A fragments are two ds_load_b128 per lane straight from row-major LDS:
//   lane L, kc: row = L%16, K = kc*32 + 8*(L/16) (+16 for upper half).
// Wave computes a 32x32 output patch = 2x2 WMMA tiles.
// ---------------------------------------------------------------------------
__global__ __launch_bounds__(256) void gemm_bf16_kernel(const bf16_t* __restrict__ A,
                                                        const bf16_t* __restrict__ Wp,
                                                        const float* __restrict__ bias,
                                                        const float* __restrict__ addend,
                                                        float* __restrict__ out,
                                                        int act) {
    __shared__ bf16_t ldsA[2][128 * 64];       // 2 x 16 KB, row-major [row][k]
    const int tid = threadIdx.x;
    const int lane = tid & 31, wave = tid >> 5;
    const int mpair = wave & 3;                // rows 32*mpair in tile
    const int npair = wave >> 2;               // cols 32*npair in tile
    const int rowBase = blockIdx.x * 128;
    const int ntileBase = blockIdx.y * 4;
    const bf16_t* Abase = A + (size_t)rowBase * DD;

    const int mrow = lane & 15;                // fragment row within 16
    const int khi = (lane >> 4) << 3;          // +8 K for upper half-wave

    v8f acc[2][2] = {};

    if (wave == 0) {
        tdm_load_tileA((unsigned)(uintptr_t)&ldsA[0][0], Abase);
        __builtin_amdgcn_s_wait_tensorcnt(0);
    }
    __syncthreads();

    for (int chunk = 0; chunk < 8; ++chunk) {  // 64 K per chunk
        const int buf = chunk & 1;
        if (wave == 0 && chunk < 7)
            tdm_load_tileA((unsigned)(uintptr_t)&ldsA[buf ^ 1][0],
                           Abase + (chunk + 1) * 64);
        // ---- 2 K-steps of WMMA from ldsA[buf] ----
#pragma unroll
        for (int kcl = 0; kcl < 2; ++kcl) {
            const int kcg = chunk * 2 + kcl;
            ABFrag a[2], b[2];
#pragma unroll
            for (int mi = 0; mi < 2; ++mi) {
                int m = (mpair * 2 + mi) * 16 + mrow;
                const bf16_t* pa = &ldsA[buf][m * 64 + kcl * 32 + khi];
                a[mi].u[0] = *(const u4v*)pa;          // K +0..7
                a[mi].u[1] = *(const u4v*)(pa + 16);   // K +16..23
            }
#pragma unroll
            for (int ni = 0; ni < 2; ++ni) {
                int ntile = ntileBase + npair * 2 + ni;
                const bf16_t* pb = Wp + ((((ntile * 16 + kcg) * 32 + lane)) << 4);
                b[ni].u[0] = ((const u4v*)pb)[0];
                b[ni].u[1] = ((const u4v*)pb)[1];
            }
#pragma unroll
            for (int mi = 0; mi < 2; ++mi)
#pragma unroll
                for (int ni = 0; ni < 2; ++ni)
                    acc[mi][ni] = wmma_bf16(a[mi].v, b[ni].v, acc[mi][ni]);
        }
        if (wave == 0 && chunk < 7) __builtin_amdgcn_s_wait_tensorcnt(0);
        __syncthreads();
    }

    // ---- epilogue ----
#pragma unroll
    for (int mi = 0; mi < 2; ++mi) {
#pragma unroll
        for (int ni = 0; ni < 2; ++ni) {
            int col = blockIdx.y * 64 + npair * 32 + ni * 16 + (lane & 15);
            int rBase = rowBase + mpair * 32 + mi * 16 + ((lane >> 4) << 3);
            float bcol = bias ? bias[col] : 0.f;
#pragma unroll
            for (int r = 0; r < 8; ++r) {
                size_t idx = (size_t)(rBase + r) * DD + col;
                float v = acc[mi][ni][r] + bcol;
                if (addend) v += addend[idx];
                if (act == 1) v = tanhf(v);
                out[idx] = v;
            }
        }
    }
}

// ---------------------------------------------------------------------------
// Sequential LRU scan. Grid = 2 blocks (16 batch rows each), 512 thr = 16 waves.
// Wave w owns output cols [32w, 32w+32). h lives in WMMA accumulator layout in
// registers; bf16 h is round-tripped through double-buffered row-major LDS
// ([16 rows][512 cols]); wave w writes its 32 columns. One barrier per step.
// ---------------------------------------------------------------------------
__global__ __launch_bounds__(512) void lru_scan_kernel(const float* __restrict__ nh,
                                                       const float* __restrict__ ifb,
                                                       const bf16_t* __restrict__ Whfp,
                                                       const float* __restrict__ bhf,
                                                       bf16_t* __restrict__ hout) {
    __shared__ bf16_t ldsH[2][16 * DD];        // 2 x 16 KB, [row][k]
    const int tid = threadIdx.x;
    const int lane = tid & 31, w = tid >> 5;   // wave 0..15
    const int bgroup = blockIdx.x;             // 0..1
    const int colLo = lane & 15;
    const int rOff = (lane >> 4) << 3;         // +8 for hi half-wave
    const int mrow = lane & 15;
    const int khi = (lane >> 4) << 3;

    // zero read-slot 0 (t=0 => h=0)
    for (int i = tid; i < 1024; i += 512) ((u4v*)(&ldsH[0][0]))[i] = (u4v)0u;
    __syncthreads();

    v8f h[2] = {};
    float bc[2];
#pragma unroll
    for (int j = 0; j < 2; ++j) bc[j] = bhf[32 * w + 16 * j + colLo];

    for (int t = 0; t < TT; ++t) {
        const int rd = t & 1, wr = rd ^ 1;
        v8f s[2] = {};
#pragma unroll
        for (int kc = 0; kc < 16; ++kc) {
            ABFrag a;
            const bf16_t* pa = &ldsH[rd][mrow * DD + kc * 32 + khi];
            a.u[0] = *(const u4v*)pa;
            a.u[1] = *(const u4v*)(pa + 16);
#pragma unroll
            for (int j = 0; j < 2; ++j) {
                int ntile = 2 * w + j;
                const bf16_t* pb = Whfp + (((ntile * 16 + kc) * 32 + lane) << 4);
                ABFrag b;
                b.u[0] = ((const u4v*)pb)[0];
                b.u[1] = ((const u4v*)pb)[1];
                s[j] = wmma_bf16(a.v, b.v, s[j]);
            }
        }
        // gate + lerp + write-out + repack h for next step
#pragma unroll
        for (int j = 0; j < 2; ++j) {
            int col = 32 * w + 16 * j + colLo;
#pragma unroll
            for (int r = 0; r < 8; ++r) {
                int rowl = rOff + r;                       // 0..15
                size_t g_idx = ((size_t)(bgroup * 16 + rowl) * TT + t) * DD + col;
                float gte = 1.f / (1.f + __expf(-(s[j][r] + bc[j] + ifb[g_idx])));
                float hv = h[j][r];
                hv = hv + gte * (nh[g_idx] - hv);
                h[j][r] = hv;
                bf16_t hb = f2bf(hv);
                hout[g_idx] = hb;
                ldsH[wr][rowl * DD + col] = hb;
            }
        }
        __syncthreads();
    }
}

// ---------------------------------------------------------------------------
// Final fused: out = xn + sigmoid(s)*(nh_g - xn) + x, xn = x*rstd*(gamma+1)
// ---------------------------------------------------------------------------
__global__ __launch_bounds__(256) void final_kernel(const float* __restrict__ x,
                                                    const float* __restrict__ gamma,
                                                    const float* __restrict__ rstd,
                                                    const float* __restrict__ s,
                                                    const float* __restrict__ nhg,
                                                    float* __restrict__ out) {
    size_t i4 = ((size_t)blockIdx.x * 256 + threadIdx.x) * 4;
    int col = (int)(i4 & 511);
    float sc = rstd[i4 >> 9];
    f4v xv = *(const f4v*)(x + i4);
    f4v sv = *(const f4v*)(s + i4);
    f4v nv = *(const f4v*)(nhg + i4);
    f4v res;
#pragma unroll
    for (int c = 0; c < 4; ++c) {
        float xn = xv[c] * sc * (gamma[col + c] + 1.f);
        float g = 1.f / (1.f + __expf(-sv[c]));
        res[c] = xn + g * (nv[c] - xn) + xv[c];
    }
    *(f4v*)(out + i4) = res;
}

// ---------------------------------------------------------------------------
extern "C" void kernel_launch(void* const* d_in, const int* in_sizes, int n_in,
                              void* d_out, int out_size, void* d_ws, size_t ws_size,
                              hipStream_t stream) {
    (void)in_sizes; (void)n_in; (void)out_size; (void)ws_size;
    const float* x     = (const float*)d_in[0];
    const float* gamma = (const float*)d_in[1];
    const float* Wsrc[9] = {
        (const float*)d_in[6],  (const float*)d_in[7],  (const float*)d_in[8],   // l0: Wn, Wif, Whf
        (const float*)d_in[10], (const float*)d_in[11], (const float*)d_in[12],  // l1: Wn, Wif, Whf
        (const float*)d_in[2],  (const float*)d_in[3],  (const float*)d_in[4],   // g:  Wn, Wif, Whf
    };
    const float* l0_bhf = (const float*)d_in[9];
    const float* l1_bhf = (const float*)d_in[13];
    const float* g_bhf  = (const float*)d_in[5];

    char* ws = (char*)d_ws;
    float*  nh   = (float*)(ws + 0);                       // 134217728 B
    float*  ifb  = (float*)(ws + 134217728u);              // 134217728 B
    bf16_t* xnb  = (bf16_t*)(ws + 268435456u);             //  67108864 B
    bf16_t* curb = (bf16_t*)(ws + 335544320u);             //  67108864 B
    float*  rstd = (float*)(ws + 402653184u);              //    262144 B
    bf16_t* wp0  = (bf16_t*)(ws + 402915328u);             // 9 x 524288 B
    bf16_t* wp[9];
    for (int i = 0; i < 9; ++i) wp[i] = wp0 + (size_t)i * (DD * DD);

    // 1) weight convert + fragment-pack (bf16)
    for (int i = 0; i < 9; ++i)
        pack_w_kernel<<<1024, 256, 0, stream>>>(Wsrc[i], wp[i]);

    // 2) rmsnorm -> bf16 xn + per-row scale
    rmsnorm_kernel<<<RTOT / 8, 256, 0, stream>>>(x, gamma, xnb, rstd);

    dim3 ggrid(RTOT / 128, DD / 64);

    // 3) layer 0 precompute + scan
    gemm_bf16_kernel<<<ggrid, 256, 0, stream>>>(xnb, wp[0], nullptr, nullptr, nh, 1);
    gemm_bf16_kernel<<<ggrid, 256, 0, stream>>>(xnb, wp[1], nullptr, nullptr, ifb, 0);
    lru_scan_kernel<<<2, 512, 0, stream>>>(nh, ifb, wp[2], l0_bhf, curb);

    // 4) layer 1 precompute + scan
    gemm_bf16_kernel<<<ggrid, 256, 0, stream>>>(curb, wp[3], nullptr, nullptr, nh, 1);
    gemm_bf16_kernel<<<ggrid, 256, 0, stream>>>(curb, wp[4], nullptr, nullptr, ifb, 0);
    lru_scan_kernel<<<2, 512, 0, stream>>>(nh, ifb, wp[5], l1_bhf, curb);

    // 5) gate cell: nh_g = tanh(h1@gWn); s = xn@gWhf + gbhf + h1@gWif
    gemm_bf16_kernel<<<ggrid, 256, 0, stream>>>(curb, wp[6], nullptr, nullptr, nh, 1);
    gemm_bf16_kernel<<<ggrid, 256, 0, stream>>>(xnb,  wp[8], g_bhf,   nullptr, ifb, 0);
    gemm_bf16_kernel<<<ggrid, 256, 0, stream>>>(curb, wp[7], nullptr, ifb,     ifb, 0);

    // 6) out = xn + sigmoid(s)*(nh_g - xn) + x
    final_kernel<<<(RTOT * DD) / (256 * 4), 256, 0, stream>>>(x, gamma, rstd, ifb, nh,
                                                              (float*)d_out);
}
